// MLP5_59751585022233
// MI455X (gfx1250) — matrix operand order repaired and checked by reading, not compiled, not run on hardware
//
#include <hip/hip_runtime.h>
#include <hip/hip_bf16.h>

typedef __attribute__((ext_vector_type(16))) __bf16 v16bf;
typedef __attribute__((ext_vector_type(8)))  __bf16 v8bf;
typedef __attribute__((ext_vector_type(8)))  float  v8f;

#define BB    256
#define LL    100
#define D0    500
#define D1    250
#define D2    250
#define XDIM  1000
#define H1DIM 200
#define H2DIM 1000
#define KPAD  1024
#define NPIX  240000

// ---------------------------------------------------------------------------
// Kernel 1: per-bucket masked embedding-bag mean -> x [256,1000] (fp32, ws)
// ---------------------------------------------------------------------------
__global__ __launch_bounds__(512) void mlp5_embed_kernel(
    const int* __restrict__ songs_bucket, const int* __restrict__ songs_id,
    const float* __restrict__ W100, const float* __restrict__ W10_100,
    const float* __restrict__ W0_10, float* __restrict__ x)
{
    const int b = blockIdx.x;
    const int t = threadIdx.x;
    float s0 = 0.f, s1 = 0.f, s2 = 0.f;
    int c0 = 0, c1 = 0, c2 = 0;
    for (int l = 0; l < LL; ++l) {
        const int bk = songs_bucket[b * LL + l];
        const int id = songs_id[b * LL + l];
        if (bk == 0) {
            ++c0;
            if (t < D0) s0 += W100[(size_t)id * D0 + t];
        } else if (bk == 1) {
            ++c1;
            if (t < D1) s1 += W10_100[(size_t)id * D1 + t];
        } else {
            ++c2;
            if (t < D2) s2 += W0_10[(size_t)id * D2 + t];
        }
    }
    float* xr = x + (size_t)b * XDIM;
    if (t < D0) xr[t] = c0 ? s0 / (float)c0 : 0.f;
    if (t < D1) {
        xr[D0 + t]      = c1 ? s1 / (float)c1 : 0.f;
        xr[D0 + D1 + t] = c2 ? s2 / (float)c2 : 0.f;
    }
}

// ---------------------------------------------------------------------------
// Kernel 2: h1 = relu(x @ fc1_w.T + fc1_b)   [256,200], thread-per-output
// ---------------------------------------------------------------------------
__global__ __launch_bounds__(256) void mlp5_fc1_kernel(
    const float* __restrict__ x, const float* __restrict__ fc1_w,
    const float* __restrict__ fc1_b, float* __restrict__ h1)
{
    const int o = blockIdx.x * blockDim.x + threadIdx.x;   // [0, 256*200)
    const int b = o / H1DIM;
    const int j = o % H1DIM;
    const float4* xp = (const float4*)(x + (size_t)b * XDIM);
    const float4* wp = (const float4*)(fc1_w + (size_t)j * XDIM);
    float s = 0.f;
    #pragma unroll 4
    for (int k = 0; k < XDIM / 4; ++k) {
        const float4 a = xp[k], w = wp[k];
        s += a.x * w.x + a.y * w.y + a.z * w.z + a.w * w.w;
    }
    s += fc1_b[j];
    h1[o] = fmaxf(s, 0.f);
}

// ---------------------------------------------------------------------------
// Kernel 3: h2 = relu(h1 @ fc2_w.T + fc2_b) -> bf16, K-padded to 1024
// ---------------------------------------------------------------------------
__global__ __launch_bounds__(256) void mlp5_fc2_kernel(
    const float* __restrict__ h1, const float* __restrict__ fc2_w,
    const float* __restrict__ fc2_b, __bf16* __restrict__ h2bf)
{
    const int o = blockIdx.x * blockDim.x + threadIdx.x;   // [0, 256*1024)
    const int b = o >> 10;
    const int j = o & (KPAD - 1);
    if (j >= H2DIM) { h2bf[o] = (__bf16)0.0f; return; }
    const float4* hp = (const float4*)(h1 + (size_t)b * H1DIM);
    const float4* wp = (const float4*)(fc2_w + (size_t)j * H1DIM);
    float s = 0.f;
    #pragma unroll 5
    for (int k = 0; k < H1DIM / 4; ++k) {
        const float4 a = hp[k], w = wp[k];
        s += a.x * w.x + a.y * w.y + a.z * w.z + a.w * w.w;
    }
    s += fc2_b[j];
    h2bf[(size_t)b * KPAD + j] = (__bf16)fmaxf(s, 0.f);
}

// ---------------------------------------------------------------------------
// Kernel 4: ratings[256,240000] = h2 @ out_w.T + out_b
// V_WMMA_F32_16X16X32_BF16. Block = 128 threads (4 waves) owns a 32-wide
// N-tile (2 sub-tiles per wave) and ALL of M=256 (wave w -> M-tiles 4w..4w+3),
// so out_w streams from HBM exactly once. K loop is peeled: 30 software-
// pipelined iterations (next out_w block prefetched into VGPRs before the 8
// WMMAs of the current one), 1 final full iteration, 1 guarded 8-element tail.
// ---------------------------------------------------------------------------
struct F4x4 { float4 f[4]; };

__device__ __forceinline__ F4x4 loadB16(const float* p) {
    F4x4 r;
    const float4* q = (const float4*)p;
    #pragma unroll
    for (int i = 0; i < 4; ++i) r.f[i] = q[i];
    return r;
}

__device__ __forceinline__ v16bf cvtB16(const F4x4& s) {
    v16bf b;
    #pragma unroll
    for (int i = 0; i < 4; ++i) {
        b[4 * i + 0] = (__bf16)s.f[i].x;
        b[4 * i + 1] = (__bf16)s.f[i].y;
        b[4 * i + 2] = (__bf16)s.f[i].z;
        b[4 * i + 3] = (__bf16)s.f[i].w;
    }
    return b;
}

__device__ __forceinline__ v16bf loadA16(const __bf16* ar) {
    // A 16x32 bf16 layout: lane row m, K = e + hi*16
    const v8bf alo = *(const v8bf*)(ar);        // K = base      .. +7
    const v8bf ahi = *(const v8bf*)(ar + 8);    // K = base + 8  .. +7
    v16bf a;
    #pragma unroll
    for (int e = 0; e < 8; ++e) { a[e] = alo[e]; a[8 + e] = ahi[e]; }
    return a;
}

__global__ __launch_bounds__(128) void mlp5_outgemm_kernel(
    const __bf16* __restrict__ h2bf, const float* __restrict__ out_w,
    const float* __restrict__ out_b, float* __restrict__ ratings)
{
    const int lane = threadIdx.x & 31;
    const int wave = threadIdx.x >> 5;
    const int lo   = lane & 15;
    const int hi   = lane >> 4;

    const int n0 = blockIdx.x * 32 + lo;            // B/D column, sub-tile 0
    const int n1 = n0 + 16;                         // sub-tile 1
    const float* bptr0 = out_w + (size_t)n0 * H2DIM + hi * 16;
    const float* bptr1 = out_w + (size_t)n1 * H2DIM + hi * 16;

    // A row pointers for the 4 M-tiles this wave owns (lane -> row m_base+lo)
    const __bf16* arow = h2bf + (size_t)(wave * 64 + lo) * KPAD + hi * 16;

    v8f acc[4][2];
    #pragma unroll
    for (int mt = 0; mt < 4; ++mt) { acc[mt][0] = v8f{}; acc[mt][1] = v8f{}; }

    // prologue: fetch K-block 0 of the out_w stream
    F4x4 fb0 = loadB16(bptr0);
    F4x4 fb1 = loadB16(bptr1);

    // ---- 30 pipelined full iterations: kk = 0,32,...,928 ----
    for (int kk = 0; kk < 960; kk += 32) {
        const v16bf bv0 = cvtB16(fb0);
        const v16bf bv1 = cvtB16(fb1);
        fb0 = loadB16(bptr0 + kk + 32);             // prefetch next K-block
        fb1 = loadB16(bptr1 + kk + 32);             // (kk+32 <= 960 -> in range)
        #pragma unroll
        for (int mt = 0; mt < 4; ++mt) {
            const v16bf av = loadA16(arow + (size_t)mt * 16 * KPAD + kk);
            acc[mt][0] = __builtin_amdgcn_wmma_f32_16x16x32_bf16(
                false, av, false, bv0, (short)0, acc[mt][0], false, false);
            acc[mt][1] = __builtin_amdgcn_wmma_f32_16x16x32_bf16(
                false, av, false, bv1, (short)0, acc[mt][1], false, false);
        }
    }

    // ---- last full iteration: kk = 960 (K 960..991, in range) ----
    {
        const v16bf bv0 = cvtB16(fb0);
        const v16bf bv1 = cvtB16(fb1);
        #pragma unroll
        for (int mt = 0; mt < 4; ++mt) {
            const v16bf av = loadA16(arow + (size_t)mt * 16 * KPAD + 960);
            acc[mt][0] = __builtin_amdgcn_wmma_f32_16x16x32_bf16(
                false, av, false, bv0, (short)0, acc[mt][0], false, false);
            acc[mt][1] = __builtin_amdgcn_wmma_f32_16x16x32_bf16(
                false, av, false, bv1, (short)0, acc[mt][1], false, false);
        }
    }

    // ---- guarded tail: kk = 992 (only K 992..999 real; A is zero-padded) ----
    {
        v16bf bv0, bv1;
        #pragma unroll
        for (int e = 0; e < 16; ++e) {
            const int k = 992 + hi * 16 + e;
            const bool ok = (k < H2DIM);
            bv0[e] = (__bf16)(ok ? out_w[(size_t)n0 * H2DIM + k] : 0.0f);
            bv1[e] = (__bf16)(ok ? out_w[(size_t)n1 * H2DIM + k] : 0.0f);
        }
        #pragma unroll
        for (int mt = 0; mt < 4; ++mt) {
            const v16bf av = loadA16(arow + (size_t)mt * 16 * KPAD + 992);
            acc[mt][0] = __builtin_amdgcn_wmma_f32_16x16x32_bf16(
                false, av, false, bv0, (short)0, acc[mt][0], false, false);
            acc[mt][1] = __builtin_amdgcn_wmma_f32_16x16x32_bf16(
                false, av, false, bv1, (short)0, acc[mt][1], false, false);
        }
    }

    // ---- epilogue: D layout M = v + hi*8, N = lane&15; add bias, store ----
    const float bias0 = out_b[n0];
    const float bias1 = out_b[n1];
    #pragma unroll
    for (int mt = 0; mt < 4; ++mt) {
        const int mbase = (wave * 4 + mt) * 16 + hi * 8;
        #pragma unroll
        for (int v = 0; v < 8; ++v) {
            ratings[(size_t)(mbase + v) * NPIX + n0] = acc[mt][0][v] + bias0;
            ratings[(size_t)(mbase + v) * NPIX + n1] = acc[mt][1][v] + bias1;
        }
    }
}

// ---------------------------------------------------------------------------
extern "C" void kernel_launch(void* const* d_in, const int* in_sizes, int n_in,
                              void* d_out, int out_size, void* d_ws, size_t ws_size,
                              hipStream_t stream) {
    const int*   songs_bucket = (const int*)d_in[0];
    const int*   songs_id     = (const int*)d_in[1];
    const float* W100         = (const float*)d_in[2];
    const float* W10_100      = (const float*)d_in[3];
    const float* W0_10        = (const float*)d_in[4];
    const float* fc1_w        = (const float*)d_in[5];
    const float* fc1_b        = (const float*)d_in[6];
    const float* fc2_w        = (const float*)d_in[7];
    const float* fc2_b        = (const float*)d_in[8];
    const float* out_w        = (const float*)d_in[9];
    const float* out_b        = (const float*)d_in[10];
    float* ratings = (float*)d_out;

    // workspace layout (all 16B aligned): x | h1 | h2 (bf16, K padded to 1024)
    float*  x    = (float*)d_ws;                      // 256*1000 f32
    float*  h1   = x + (size_t)BB * XDIM;             // 256*200  f32
    __bf16* h2bf = (__bf16*)(h1 + (size_t)BB * H1DIM);// 256*1024 bf16 (~1.72MB total)

    mlp5_embed_kernel<<<BB, 512, 0, stream>>>(songs_bucket, songs_id,
                                              W100, W10_100, W0_10, x);
    mlp5_fc1_kernel<<<(BB * H1DIM) / 256, 256, 0, stream>>>(x, fc1_w, fc1_b, h1);
    mlp5_fc2_kernel<<<(BB * KPAD) / 256, 256, 0, stream>>>(h1, fc2_w, fc2_b, h2bf);
    mlp5_outgemm_kernel<<<NPIX / 32, 128, 0, stream>>>(h2bf, out_w, out_b, ratings);
}
